// SelfAttention_83004537962569
// MI455X (gfx1250) — matrix operand order repaired and checked by reading, not compiled, not run on hardware
//
#include <hip/hip_runtime.h>
#include <hip/hip_bf16.h>

// ---------------------------------------------------------------------------
// Self-attention for MI455X (gfx1250): bf16 WMMA + TDM async tensor loads.
// ---------------------------------------------------------------------------

#define B_   8
#define T_   1024
#define C_   768
#define NH_  12
#define HS_  64
#define N3C  2304
#define MTOT (B_ * T_) // 8192

typedef __attribute__((ext_vector_type(16))) __bf16   v16bf;
typedef __attribute__((ext_vector_type(8)))  __bf16   v8bf;
typedef __attribute__((ext_vector_type(8)))  float    v8f;
typedef __attribute__((ext_vector_type(4)))  unsigned v4u;
typedef __attribute__((ext_vector_type(8)))  unsigned v8u;

static __device__ __forceinline__ __bf16 f2bf(float f) {
    unsigned u = __float_as_uint(f);
    u += 0x7FFFu + ((u >> 16) & 1u); // round-to-nearest-even
    unsigned short h = (unsigned short)(u >> 16);
    return __builtin_bit_cast(__bf16, h);
}

// Build a 16-element bf16 fragment from two 16-byte LDS loads.
static __device__ __forceinline__ v16bf ld_pair(const __bf16* p0, const __bf16* p1) {
    v8bf a = *(const v8bf*)p0;
    v8bf b = *(const v8bf*)p1;
    return __builtin_shufflevector(a, b, 0,1,2,3,4,5,6,7,8,9,10,11,12,13,14,15);
}

static __device__ __forceinline__ v8f wmma_bf16(v16bf a, v16bf b, v8f c) {
    return __builtin_amdgcn_wmma_f32_16x16x32_bf16(false, a, false, b,
                                                   (short)0, c, false, false);
}

static __device__ __forceinline__ v8f zero8() {
    v8f z = {0.f, 0.f, 0.f, 0.f, 0.f, 0.f, 0.f, 0.f};
    return z;
}

// A-fragment (16x32, row-major in LDS, stride % 8 == 0):
// lane L (0..15): row=L, K 0..7 then 16..23 ; lane L+16: row=L, K 8..15 then 24..31
static __device__ __forceinline__ v16bf ldA(const __bf16* tile, int stride, int lane) {
    int r  = lane & 15;
    int kb = (lane >> 4) * 8;
    const __bf16* p = tile + r * stride + kb;
    return ld_pair(p, p + 16);
}
// B-fragment (32x16 KxN), LDS holds the tile as [N][K] (stride % 8 == 0):
// lane L (0..15): col=L, K 0..7 then 16..23 ; lane L+16: col=L, K 8..15 then 24..31
static __device__ __forceinline__ v16bf ldB(const __bf16* tileNK, int stride, int lane) {
    int n  = lane & 15;
    int kb = (lane >> 4) * 8;
    const __bf16* p = tileNK + n * stride + kb;
    return ld_pair(p, p + 16);
}

// ---------------------------------------------------------------------------
// Kernel 1: qkv = x @ Wqkv + bqkv  -> bf16 Q/K/V in [B, NH, T, 64] layout
// block tile 64(M) x 256(N), K-step 32; 8 waves in a 2x4 grid, 32x64 per wave
// ---------------------------------------------------------------------------
__global__ __launch_bounds__(256) void qkv_kernel(
    const float* __restrict__ x, const float* __restrict__ Wqkv,
    const float* __restrict__ bqkv,
    __bf16* __restrict__ qw, __bf16* __restrict__ kw, __bf16* __restrict__ vw)
{
    __shared__ __bf16 As[64 * 40];   // [m][k], stride 40
    __shared__ __bf16 Bs[256 * 40];  // [n][k], stride 40 (transposed on store)

    const int tid  = threadIdx.x;
    const int lane = tid & 31;
    const int wave = tid >> 5;
    const int wm   = wave >> 2;      // 0..1  -> 32 rows
    const int wn   = wave & 3;       // 0..3  -> 64 cols

    const int m0 = blockIdx.x * 64;
    const int n0 = blockIdx.y * 256;

    v8f acc[2][4];
#pragma unroll
    for (int mi = 0; mi < 2; ++mi)
#pragma unroll
        for (int ni = 0; ni < 4; ++ni) acc[mi][ni] = zero8();

    const int arow = tid >> 2;          // 0..63
    const int acol = (tid & 3) * 8;     // 0,8,16,24
    const int brow = tid >> 3;          // 0..31   (k)
    const int bcol = (tid & 7) * 32;    // 0..224  (n)

    for (int k0 = 0; k0 < C_; k0 += 32) {
        // A: x tile fp32 -> bf16 LDS (row-major)
        {
            const float* ap = x + (size_t)(m0 + arow) * C_ + k0 + acol;
            __builtin_prefetch(ap + 32, 0, 0);       // next K tile
            float4 a0 = *(const float4*)ap;
            float4 a1 = *(const float4*)(ap + 4);
            __bf16* as = As + arow * 40 + acol;
            as[0] = f2bf(a0.x); as[1] = f2bf(a0.y); as[2] = f2bf(a0.z); as[3] = f2bf(a0.w);
            as[4] = f2bf(a1.x); as[5] = f2bf(a1.y); as[6] = f2bf(a1.z); as[7] = f2bf(a1.w);
        }
        // B: Wqkv tile fp32 (row-major [k][n]) -> bf16 LDS transposed [n][k]
        {
            const float* bp = Wqkv + (size_t)(k0 + brow) * N3C + n0 + bcol;
            __builtin_prefetch(bp + (size_t)32 * N3C, 0, 0); // next K tile
#pragma unroll
            for (int i = 0; i < 32; i += 4) {
                float4 t = *(const float4*)(bp + i);
                Bs[(bcol + i + 0) * 40 + brow] = f2bf(t.x);
                Bs[(bcol + i + 1) * 40 + brow] = f2bf(t.y);
                Bs[(bcol + i + 2) * 40 + brow] = f2bf(t.z);
                Bs[(bcol + i + 3) * 40 + brow] = f2bf(t.w);
            }
        }
        __syncthreads();

        v16bf afrag[2];
#pragma unroll
        for (int mi = 0; mi < 2; ++mi)
            afrag[mi] = ldA(As + (wm * 32 + mi * 16) * 40, 40, lane);
#pragma unroll
        for (int ni = 0; ni < 4; ++ni) {
            v16bf bfrag = ldB(Bs + (wn * 64 + ni * 16) * 40, 40, lane);
#pragma unroll
            for (int mi = 0; mi < 2; ++mi)
                acc[mi][ni] = wmma_bf16(afrag[mi], bfrag, acc[mi][ni]);
        }
        __syncthreads();
    }

    // Epilogue: + bias, scatter bf16 into Q/K/V [B][NH][T][HS]
    const int rbase = 8 * (lane >> 4);
    const int ncol  = lane & 15;
#pragma unroll
    for (int mi = 0; mi < 2; ++mi) {
#pragma unroll
        for (int ni = 0; ni < 4; ++ni) {
            int gcol = n0 + wn * 64 + ni * 16 + ncol;
            int mat  = gcol / C_;
            int rem  = gcol - mat * C_;
            int h    = rem >> 6;
            int d    = rem & 63;
            float bias = bqkv[gcol];
            __bf16* dst = (mat == 0) ? qw : ((mat == 1) ? kw : vw);
#pragma unroll
            for (int r = 0; r < 8; ++r) {
                int grow = m0 + wm * 32 + mi * 16 + rbase + r;
                int b = grow >> 10;
                int t = grow & 1023;
                dst[((size_t)(b * NH_ + h) * T_ + t) * HS_ + d] =
                    f2bf(acc[mi][ni][r] + bias);
            }
        }
    }
}

// ---------------------------------------------------------------------------
// Kernel 2: causal flash attention.  One workgroup per (b, h, 128-query tile);
// wave w owns 16 query rows.  K tile staged by the Tensor Data Mover (TDM),
// key blocks of 64, online softmax, fp32 state.
// ---------------------------------------------------------------------------
__global__ __launch_bounds__(256) void attn_kernel(
    const __bf16* __restrict__ qw, const __bf16* __restrict__ kw,
    const __bf16* __restrict__ vw, __bf16* __restrict__ yw)
{
    const int qt = blockIdx.x & 7;              // T/128 = 8 tiles
    const int h  = (blockIdx.x >> 3) % NH_;
    const int b  = blockIdx.x / (8 * NH_);
    const int q0 = qt * 128;

    __shared__ __bf16 Qs[128 * 72];     // [q][d]
    __shared__ __bf16 Ks[64 * 72];      // [key][d]   (B operand for Q@K^T) - TDM
    __shared__ __bf16 Vs[64 * 72];      // [d][key]   (B operand for P@V)
    __shared__ __bf16 Ps[8][16 * 72];   // per-wave P tile [q][key]

    const int tid  = threadIdx.x;
    const int lane = tid & 31;
    const int wave = tid >> 5;

    const __bf16* Qg = qw + ((size_t)(b * NH_ + h) * T_ + q0) * HS_;
    const __bf16* Kg = kw + (size_t)(b * NH_ + h) * T_ * HS_;
    const __bf16* Vg = vw + (size_t)(b * NH_ + h) * T_ * HS_;

    // Load the 128x64 Q tile (bf16, vectorized 16B)
    {
        int row = tid >> 1;
        int col = (tid & 1) * 32;
        const v8bf* src = (const v8bf*)(Qg + (size_t)row * HS_ + col);
        v8bf* dst = (v8bf*)(Qs + row * 72 + col);
        dst[0] = src[0]; dst[1] = src[1]; dst[2] = src[2]; dst[3] = src[3];
    }

    // TDM descriptor parts that do not change across key blocks (wave-uniform).
    const unsigned lds_k = (unsigned)(size_t)(void*)&Ks[0];

    float mrow[8], lrow[8];
    v8f o[4];
#pragma unroll
    for (int r = 0; r < 8; ++r) { mrow[r] = -1e30f; lrow[r] = 0.f; }
#pragma unroll
    for (int nf = 0; nf < 4; ++nf) o[nf] = zero8();

    const float scale = 0.125f;             // 1/sqrt(64)
    const int   rbase = 8 * (lane >> 4);
    const int   nkb   = (q0 + 128) >> 6;    // causal: qt*2 + 2 key blocks

    for (int kb = 0; kb < nkb; ++kb) {
        __syncthreads();    // previous iteration's readers done with Ks/Vs

        // --- K tile via Tensor Data Mover: 64x64 bf16 tile, row stride 64,
        //     LDS padding 4 dwords every 32 dwords -> stride-72 layout.
        //     Issued by wave 0 only (TDM ignores EXEC -> scalar branch).
        if (__builtin_amdgcn_readfirstlane(tid) < 32) {
            unsigned long long ga =
                (unsigned long long)(const void*)(Kg + (size_t)(kb * 64) * HS_);
            unsigned ga_lo = (unsigned)__builtin_amdgcn_readfirstlane((int)ga);
            unsigned ga_hi = (unsigned)__builtin_amdgcn_readfirstlane((int)(ga >> 32));
            unsigned ldsa  = (unsigned)__builtin_amdgcn_readfirstlane((int)lds_k);
            v4u g0;
            g0[0] = 1u;                                   // count=1 (valid user D#)
            g0[1] = ldsa;                                 // lds_addr
            g0[2] = ga_lo;                                // global_addr[31:0]
            g0[3] = (ga_hi & 0x01FFFFFFu) | (2u << 30);   // global_addr[56:32]|type=2
            v8u g1;
            g1[0] = (1u << 16)      // data_size = 2 bytes
                  | (1u << 20)      // pad_enable
                  | (4u << 22)      // pad_interval = 32 dwords (64 bf16)
                  | (3u << 25);     // pad_amount   = 4 dwords (16 bytes)
            g1[1] = (64u   << 16);  // tensor_dim0 = 64 elements
            g1[2] = (1024u << 16);  // tensor_dim1 = T keys
            g1[3] = (64u   << 16);  // tile_dim0 = 64
            g1[4] = 64u;            // tile_dim1 = 64 keys (tile_dim2 = 0)
            g1[5] = 64u;            // tensor_dim0_stride = 64 elements
            g1[6] = 0u;
            g1[7] = 0u;
            asm volatile("tensor_load_to_lds %0, %1" :: "s"(g0), "s"(g1) : "memory");
            __builtin_amdgcn_s_wait_tensorcnt(0);
        }

        // --- V block, transposed into [d][key] (regular path; TDM can't transpose)
        {
            int key  = tid >> 2;
            int dseg = (tid & 3) * 16;
            const __bf16* vs = Vg + (size_t)(kb * 64 + key) * HS_ + dseg;
#pragma unroll
            for (int i = 0; i < 16; ++i)
                Vs[(dseg + i) * 72 + key] = vs[i];
        }
        __syncthreads();

        // S = Q @ K^T  (16 q-rows x 64 keys per wave)
        v8f s[4];
#pragma unroll
        for (int nf = 0; nf < 4; ++nf) s[nf] = zero8();
        v16bf aq[2];
#pragma unroll
        for (int ks2 = 0; ks2 < 2; ++ks2)
            aq[ks2] = ldA(Qs + (wave * 16) * 72 + ks2 * 32, 72, lane);
#pragma unroll
        for (int nf = 0; nf < 4; ++nf)
#pragma unroll
            for (int ks2 = 0; ks2 < 2; ++ks2) {
                v16bf bk = ldB(Ks + (nf * 16) * 72 + ks2 * 32, 72, lane);
                s[nf] = wmma_bf16(aq[ks2], bk, s[nf]);
            }

        // scale + causal mask
#pragma unroll
        for (int nf = 0; nf < 4; ++nf) {
            int key = kb * 64 + nf * 16 + (lane & 15);
#pragma unroll
            for (int r = 0; r < 8; ++r) {
                int q = q0 + wave * 16 + rbase + r;
                float v = s[nf][r] * scale;
                s[nf][r] = (key <= q) ? v : -1e30f;
            }
        }

        // online softmax: row reductions across the 16-lane half-groups
#pragma unroll
        for (int r = 0; r < 8; ++r) {
            float mx = fmaxf(fmaxf(s[0][r], s[1][r]), fmaxf(s[2][r], s[3][r]));
            mx = fmaxf(mx, __shfl_xor(mx, 1, 32));
            mx = fmaxf(mx, __shfl_xor(mx, 2, 32));
            mx = fmaxf(mx, __shfl_xor(mx, 4, 32));
            mx = fmaxf(mx, __shfl_xor(mx, 8, 32));
            float mnew  = fmaxf(mrow[r], mx);
            float alpha = __expf(mrow[r] - mnew);
            mrow[r] = mnew;
            float sum = 0.f;
#pragma unroll
            for (int nf = 0; nf < 4; ++nf) {
                float p = __expf(s[nf][r] - mnew);
                s[nf][r] = p;
                sum += p;
            }
            sum += __shfl_xor(sum, 1, 32);
            sum += __shfl_xor(sum, 2, 32);
            sum += __shfl_xor(sum, 4, 32);
            sum += __shfl_xor(sum, 8, 32);
            lrow[r] = lrow[r] * alpha + sum;
#pragma unroll
            for (int nf = 0; nf < 4; ++nf) o[nf][r] *= alpha;
        }

        // P -> per-wave LDS (C-fragment layout -> A-fragment layout)
        __bf16* pw = &Ps[wave][0];
#pragma unroll
        for (int nf = 0; nf < 4; ++nf)
#pragma unroll
            for (int r = 0; r < 8; ++r)
                pw[(rbase + r) * 72 + nf * 16 + (lane & 15)] = f2bf(s[nf][r]);
        asm volatile("s_wait_dscnt 0" ::: "memory"); // same-wave LDS RAW

        // O += P @ V   (N = head dim, K = keys; Vs is [d][key])
        v16bf ap[2];
#pragma unroll
        for (int ks2 = 0; ks2 < 2; ++ks2)
            ap[ks2] = ldA(pw + ks2 * 32, 72, lane);
#pragma unroll
        for (int nf = 0; nf < 4; ++nf)
#pragma unroll
            for (int ks2 = 0; ks2 < 2; ++ks2) {
                v16bf bv = ldB(Vs + (nf * 16) * 72 + ks2 * 32, 72, lane);
                o[nf] = wmma_bf16(ap[ks2], bv, o[nf]);
            }
    }

    // Epilogue: normalize and store bf16 y [B][T][C]
    __bf16* yrow = yw + ((size_t)(b * T_) + q0 + wave * 16) * C_ + h * HS_;
#pragma unroll
    for (int nf = 0; nf < 4; ++nf)
#pragma unroll
        for (int r = 0; r < 8; ++r) {
            float val = o[nf][r] / lrow[r];
            yrow[(size_t)(rbase + r) * C_ + nf * 16 + (lane & 15)] = f2bf(val);
        }
}

// ---------------------------------------------------------------------------
// Kernel 3: out = y @ Wproj + bproj  (bf16 A in ws, fp32 W converted, fp32 out)
// block tile 64 x 256, wave tile 32 x 64
// ---------------------------------------------------------------------------
__global__ __launch_bounds__(256) void proj_kernel(
    const __bf16* __restrict__ y, const float* __restrict__ Wproj,
    const float* __restrict__ bproj, float* __restrict__ out)
{
    __shared__ __bf16 As[64 * 40];   // [m][k]
    __shared__ __bf16 Bs[256 * 40];  // [n][k]

    const int tid  = threadIdx.x;
    const int lane = tid & 31;
    const int wave = tid >> 5;
    const int wm   = wave >> 2;
    const int wn   = wave & 3;

    const int m0 = blockIdx.x * 64;
    const int n0 = blockIdx.y * 256;

    v8f acc[2][4];
#pragma unroll
    for (int mi = 0; mi < 2; ++mi)
#pragma unroll
        for (int ni = 0; ni < 4; ++ni) acc[mi][ni] = zero8();

    const int arow = tid >> 2;
    const int acol = (tid & 3) * 8;
    const int brow = tid >> 3;          // 0..31
    const int bcol = (tid & 7) * 32;    // 0..224

    for (int k0 = 0; k0 < C_; k0 += 32) {
        // A: bf16 copy (16B)
        {
            const __bf16* ap = y + (size_t)(m0 + arow) * C_ + k0 + acol;
            __builtin_prefetch(ap + 32, 0, 0);
            *(v8bf*)(As + arow * 40 + acol) = *(const v8bf*)ap;
        }
        // B: fp32 -> bf16 transposed
        {
            const float* bp = Wproj + (size_t)(k0 + brow) * C_ + n0 + bcol;
            __builtin_prefetch(bp + (size_t)32 * C_, 0, 0);
#pragma unroll
            for (int i = 0; i < 32; i += 4) {
                float4 t = *(const float4*)(bp + i);
                Bs[(bcol + i + 0) * 40 + brow] = f2bf(t.x);
                Bs[(bcol + i + 1) * 40 + brow] = f2bf(t.y);
                Bs[(bcol + i + 2) * 40 + brow] = f2bf(t.z);
                Bs[(bcol + i + 3) * 40 + brow] = f2bf(t.w);
            }
        }
        __syncthreads();

        v16bf afrag[2];
#pragma unroll
        for (int mi = 0; mi < 2; ++mi)
            afrag[mi] = ldA(As + (wm * 32 + mi * 16) * 40, 40, lane);
#pragma unroll
        for (int ni = 0; ni < 4; ++ni) {
            v16bf bfrag = ldB(Bs + (wn * 64 + ni * 16) * 40, 40, lane);
#pragma unroll
            for (int mi = 0; mi < 2; ++mi)
                acc[mi][ni] = wmma_bf16(afrag[mi], bfrag, acc[mi][ni]);
        }
        __syncthreads();
    }

    const int rbase = 8 * (lane >> 4);
    const int ncol  = lane & 15;
#pragma unroll
    for (int mi = 0; mi < 2; ++mi)
#pragma unroll
        for (int ni = 0; ni < 4; ++ni) {
            int gcol = n0 + wn * 64 + ni * 16 + ncol;
            float bias = bproj[gcol];
#pragma unroll
            for (int r = 0; r < 8; ++r) {
                int grow = m0 + wm * 32 + mi * 16 + rbase + r;
                out[(size_t)grow * C_ + gcol] = acc[mi][ni][r] + bias;
            }
        }
}

// ---------------------------------------------------------------------------
extern "C" void kernel_launch(void* const* d_in, const int* in_sizes, int n_in,
                              void* d_out, int out_size, void* d_ws, size_t ws_size,
                              hipStream_t stream) {
    const float* x     = (const float*)d_in[0];
    const float* Wqkv  = (const float*)d_in[1];
    const float* bqkv  = (const float*)d_in[2];
    const float* Wproj = (const float*)d_in[3];
    const float* bproj = (const float*)d_in[4];
    float* out = (float*)d_out;

    const size_t per_bytes = (size_t)B_ * NH_ * T_ * HS_ * sizeof(unsigned short);
    char* ws = (char*)d_ws;
    __bf16* qws = (__bf16*)(ws + 0 * per_bytes);
    __bf16* kws = (__bf16*)(ws + 1 * per_bytes);
    __bf16* vws = (__bf16*)(ws + 2 * per_bytes);
    __bf16* yws = (__bf16*)(ws + 3 * per_bytes);

    dim3 blk(256);
    qkv_kernel<<<dim3(MTOT / 64, N3C / 256), blk, 0, stream>>>(
        x, Wqkv, bqkv, qws, kws, vws);
    attn_kernel<<<dim3(B_ * NH_ * (T_ / 128)), blk, 0, stream>>>(
        qws, kws, vws, yws);
    proj_kernel<<<dim3(MTOT / 64, C_ / 256), blk, 0, stream>>>(
        yws, Wproj, bproj, out);
}
